// MLP8bit_43267500540295
// MI455X (gfx1250) — 8-bit matrix operand order repaired and checked by reading, compiled, not run on hardware
//
#include <hip/hip_runtime.h>
#include <cstdint>

typedef __attribute__((ext_vector_type(8))) int          v8i;
typedef __attribute__((ext_vector_type(8))) float        v8f;
typedef __attribute__((ext_vector_type(2))) float        v2f;
typedef __attribute__((ext_vector_type(4))) int          v4i;
typedef __attribute__((ext_vector_type(4))) unsigned int v4u;

#define OUTLIER_THRESH 3.0f

#if __has_builtin(__builtin_amdgcn_tensor_load_to_lds)
#define MLP8_HAS_TDM 1
#else
#define MLP8_HAS_TDM 0
#endif

#if __has_include(<hip/amd_detail/amd_gfx1250_TDM.h>)
#define MLP8_TDM_6ARG 1
#else
#define MLP8_TDM_6ARG 0
#endif

#if __has_builtin(__builtin_amdgcn_global_load_async_to_lds_b128)
#define MLP8_HAS_ASYNC128 1
#else
#define MLP8_HAS_ASYNC128 0
#endif

#if __has_builtin(__builtin_amdgcn_global_load_async_to_lds_b32)
#define MLP8_HAS_ASYNC32 1
#else
#define MLP8_HAS_ASYNC32 0
#endif

typedef __attribute__((address_space(1))) int mlp8_gint;
typedef __attribute__((address_space(3))) int mlp8_lint;
#define MLP8_AS1I(p) ((mlp8_gint*)(p))
#define MLP8_AS3I(p) ((mlp8_lint*)(p))
#define MLP8_AS3V(p) ((__attribute__((address_space(3))) void*)(p))

__global__ void mlp8_colmax_mask(const float* __restrict__ X,
                                 unsigned char* __restrict__ mask,
                                 int M, int K) {
  int k = blockIdx.x * blockDim.x + threadIdx.x;
  if (k >= K) return;
  float cm = 0.0f;
  const float* p = X + k;
  for (int m = 0; m < M; ++m) cm = fmaxf(cm, fabsf(p[(size_t)m * K]));
  mask[k] = (cm > OUTLIER_THRESH) ? 1 : 0;
}

__global__ void mlp8_compact_outliers(const unsigned char* __restrict__ mask,
                                      int* __restrict__ oidx,
                                      int* __restrict__ n_out, int K) {
  if (blockIdx.x == 0 && threadIdx.x == 0) {
    int c = 0;
    for (int k = 0; k < K; ++k)
      if (mask[k]) oidx[c++] = k;
    *n_out = c;
  }
}

__global__ void mlp8_quant_rows(const float* __restrict__ X,
                                const unsigned char* __restrict__ mask,
                                signed char* __restrict__ Q,
                                float* __restrict__ S,
                                int K, int use_mask) {
  __shared__ float red[256];
  const int row = blockIdx.x;
  const float* xr = X + (size_t)row * K;

  float amax = 0.0f;
  for (int k = threadIdx.x; k < K; k += blockDim.x) {
    float v = xr[k];
    if (use_mask && mask[k]) v = 0.0f;
    amax = fmaxf(amax, fabsf(v));
  }
  red[threadIdx.x] = amax;
  __syncthreads();
  for (int s = blockDim.x >> 1; s > 0; s >>= 1) {
    if (threadIdx.x < s) red[threadIdx.x] = fmaxf(red[threadIdx.x], red[threadIdx.x + s]);
    __syncthreads();
  }
  const float scale = fmaxf(red[0] / 127.0f, 1e-8f);
  if (threadIdx.x == 0) S[row] = scale;
  const float inv = 1.0f / scale;

  signed char* qr = Q + (size_t)row * K;
  for (int k = threadIdx.x; k < K; k += blockDim.x) {
    float v = xr[k];
    if (use_mask && mask[k]) v = 0.0f;
    float q = fminf(127.0f, fmaxf(-127.0f, rintf(v * inv)));
    qr[k] = (signed char)q;
  }
}

#if MLP8_HAS_TDM
__device__ __forceinline__ void mlp8_tdm_load_tile64(unsigned lds_off,
                                                     const signed char* gptr,
                                                     int K, int rows) {
  unsigned long long ga = (unsigned long long)(uintptr_t)gptr;
  v4u g0;
  g0.x = 1u;
  g0.y = lds_off;
  g0.z = (unsigned)(ga & 0xFFFFFFFFull);
  g0.w = ((unsigned)((ga >> 32) & 0x01FFFFFFull)) | 0x80000000u;
  v8i g1;
  g1[0] = 0;
  g1[1] = (int)(((unsigned)K & 0xFFFFu) << 16);
  g1[2] = (int)((((unsigned)K >> 16) & 0xFFFFu) |
                (((unsigned)rows & 0xFFFFu) << 16));
  g1[3] = (int)((((unsigned)rows >> 16) & 0xFFFFu) |
                (64u << 16));
  g1[4] = 64;
  g1[5] = K;
  g1[6] = 0;
  g1[7] = 0;
  v4i z4 = {};
#if MLP8_TDM_6ARG
  v8i z8 = {};
  __builtin_amdgcn_tensor_load_to_lds(g0, g1, z4, z4, z8, 0);
#else
  __builtin_amdgcn_tensor_load_to_lds(g0, g1, z4, z4, 0);
#endif
}
#endif

__global__ __launch_bounds__(128)
void mlp8_gemm_int8_outlier(const signed char* __restrict__ qx,
                            const float* __restrict__ sx,
                            const signed char* __restrict__ qw,
                            const float* __restrict__ sw,
                            const float* __restrict__ Xf,
                            const float* __restrict__ Wf,
                            const int* __restrict__ oidx,
                            const int* __restrict__ n_out_p,
                            const float* __restrict__ bias,
                            float* __restrict__ Y,
                            int M, int N, int K) {
  __shared__ __align__(16) signed char qxT[64][64];
  __shared__ __align__(16) signed char qwT[64][64];
  __shared__ __align__(16) float xoT[64][68];
  __shared__ __align__(16) float woT[64][68];

  const int row0 = blockIdx.y * 64;
  const int col0 = blockIdx.x * 64;
  const int tid  = threadIdx.x;
  const int lane = tid & 31;
  const int wave = tid >> 5;

  v8i acc_i[4] = {};
  v8f acc_f[4] = {};

  for (int k0 = 0; k0 < K; k0 += 64) {
#if MLP8_HAS_TDM
    if (wave == 0) {
      unsigned qx_lds = (unsigned)(uintptr_t)MLP8_AS3V(&qxT[0][0]);
      unsigned qw_lds = (unsigned)(uintptr_t)MLP8_AS3V(&qwT[0][0]);
      mlp8_tdm_load_tile64(qx_lds, qx + (size_t)row0 * K + k0, K, M);
      mlp8_tdm_load_tile64(qw_lds, qw + (size_t)col0 * K + k0, K, N);
      __builtin_amdgcn_s_wait_tensorcnt(0);
    }
#elif MLP8_HAS_ASYNC128
    {
      const signed char* gx = qx + (size_t)row0 * K + k0;
      const signed char* gw = qw + (size_t)col0 * K + k0;
      for (int c = tid; c < 256; c += 128) {
        int r = c >> 2, q = c & 3;
        __builtin_amdgcn_global_load_async_to_lds_b128(
            MLP8_AS1I(gx + (size_t)r * K + q * 16),
            MLP8_AS3I((signed char*)qxT + c * 16), 0, 0);
        __builtin_amdgcn_global_load_async_to_lds_b128(
            MLP8_AS1I(gw + (size_t)r * K + q * 16),
            MLP8_AS3I((signed char*)qwT + c * 16), 0, 0);
      }
      asm volatile("s_wait_asynccnt 0x0" ::: "memory");
    }
#else
    {
      const int4* gx4 = (const int4*)(qx + (size_t)row0 * K + k0);
      const int4* gw4 = (const int4*)(qw + (size_t)col0 * K + k0);
      for (int c = tid; c < 256; c += 128) {
        int r = c >> 2, q = c & 3;
        ((int4*)qxT)[c] = gx4[(size_t)r * (K >> 4) + q];
        ((int4*)qwT)[c] = gw4[(size_t)r * (K >> 4) + q];
      }
    }
#endif
    if (k0 + 64 < K) {
      __builtin_prefetch(qx + (size_t)(row0 + (tid >> 1)) * K + k0 + 64, 0, 3);
      __builtin_prefetch(qw + (size_t)(col0 + (tid >> 1)) * K + k0 + 64, 0, 3);
    }
    __syncthreads();

    v8i a;
    {
      const int m    = lane & 15;
      const int koff = (lane >> 4) * 8;
      const signed char* base = &qxT[wave * 16 + m][0];
      #pragma unroll
      for (int v = 0; v < 8; ++v) {
        int kb = ((v >> 1) * 16) + ((v & 1) * 4) + koff;
        a[v] = *(const int*)(base + kb);
      }
    }
    #pragma unroll
    for (int f = 0; f < 4; ++f) {
      v8i bfrag;
      const int n     = lane & 15;
      const int khalf = (lane >> 4) * 8;
      const signed char* base = &qwT[f * 16 + n][0];
      #pragma unroll
      for (int v = 0; v < 8; ++v) {
        int kb = ((v >> 1) * 16) + ((v & 1) * 4) + khalf;
        bfrag[v] = *(const int*)(base + kb);
      }
      acc_i[f] = __builtin_amdgcn_wmma_i32_16x16x64_iu8(
           true, a,  true, bfrag, acc_i[f],
           false,  false);
    }
    __syncthreads();
  }

  const int nOut = *n_out_p;
  for (int t0 = 0; t0 < nOut; t0 += 64) {
#if MLP8_HAS_ASYNC32
    for (int i = tid; i < 4096; i += 128) {
      int r = i >> 6, t = i & 63;
      if (t0 + t < nOut) {
        int kk = oidx[t0 + t];
        __builtin_amdgcn_global_load_async_to_lds_b32(
            MLP8_AS1I(Xf + (size_t)(row0 + r) * K + kk), MLP8_AS3I(&xoT[r][t]), 0, 0);
        __builtin_amdgcn_global_load_async_to_lds_b32(
            MLP8_AS1I(Wf + (size_t)(col0 + r) * K + kk), MLP8_AS3I(&woT[r][t]), 0, 0);
      } else {
        xoT[r][t] = 0.0f;
        woT[r][t] = 0.0f;
      }
    }
    asm volatile("s_wait_asynccnt 0x0" ::: "memory");
#else
    for (int i = tid; i < 4096; i += 128) {
      int r = i >> 6, t = i & 63;
      int kk = (t0 + t < nOut) ? oidx[t0 + t] : -1;
      xoT[r][t] = (kk >= 0) ? Xf[(size_t)(row0 + r) * K + kk] : 0.0f;
      woT[r][t] = (kk >= 0) ? Wf[(size_t)(col0 + r) * K + kk] : 0.0f;
    }
#endif
    __syncthreads();

    for (int s = 0; s < 16; ++s) {
      const int kk = s * 4 + (lane >> 4) * 2;
#if __has_builtin(__builtin_amdgcn_wmma_f32_16x16x4_f32)
      v2f a2;
      {
        const int m = lane & 15;
        a2[0] = xoT[wave * 16 + m][kk];
        a2[1] = xoT[wave * 16 + m][kk + 1];
      }
      #pragma unroll
      for (int f = 0; f < 4; ++f) {
        v2f b2;
        const int n = lane & 15;
        b2[0] = woT[f * 16 + n][kk];
        b2[1] = woT[f * 16 + n][kk + 1];
        acc_f[f] = __builtin_amdgcn_wmma_f32_16x16x4_f32(
            false, a2, false, b2, (short)0, acc_f[f], false, false);
      }
#else
      const int n  = lane & 15;
      const int hi = lane >> 4;
      (void)kk;
      #pragma unroll
      for (int f = 0; f < 4; ++f) {
        #pragma unroll
        for (int v = 0; v < 8; ++v) {
          float sum = 0.0f;
          #pragma unroll
          for (int u = 0; u < 4; ++u)
            sum += xoT[wave * 16 + v + 8 * hi][s * 4 + u] * woT[f * 16 + n][s * 4 + u];
          acc_f[f][v] += sum;
        }
      }
#endif
    }
    __syncthreads();
  }

  const int n_ = lane & 15;
  const int hi = lane >> 4;
  #pragma unroll
  for (int f = 0; f < 4; ++f) {
    const int col = col0 + f * 16 + n_;
    const float swv = sw[col];
    const float bv  = bias[col];
    #pragma unroll
    for (int v = 0; v < 8; ++v) {
      const int row = row0 + wave * 16 + v + 8 * hi;
      Y[(size_t)row * N + col] =
          sx[row] * swv * (float)acc_i[f][v] + acc_f[f][v] + bv;
    }
  }
}

static void run_layer(const float* in, const float* Wp, const float* bp,
                      float* out, int M, int N, int K,
                      signed char* qx, signed char* qw, float* sxb, float* swb,
                      unsigned char* mask, int* oidx, int* n_out,
                      hipStream_t stream) {
  mlp8_colmax_mask<<<(K + 255) / 256, 256, 0, stream>>>(in, mask, M, K);
  mlp8_compact_outliers<<<1, 1, 0, stream>>>(mask, oidx, n_out, K);
  mlp8_quant_rows<<<M, 256, 0, stream>>>(in, mask, qx, sxb, K, 1);
  mlp8_quant_rows<<<N, 256, 0, stream>>>(Wp, mask, qw, swb, K, 0);
  dim3 grid(N / 64, M / 64);
  mlp8_gemm_int8_outlier<<<grid, 128, 0, stream>>>(
      qx, sxb, qw, swb, in, Wp, oidx, n_out, bp, out, M, N, K);
}

extern "C" void kernel_launch(void* const* d_in, const int* in_sizes, int n_in,
                              void* d_out, int out_size, void* d_ws, size_t ws_size,
                              hipStream_t stream) {
  (void)in_sizes; (void)n_in; (void)out_size; (void)ws_size;
  const int M  = 4 * 2048;
  const int D1 = 4096;
  const int D2 = 16384;

  const float* x  = (const float*)d_in[0];
  const float* W1 = (const float*)d_in[1];
  const float* b1 = (const float*)d_in[2];
  const float* W2 = (const float*)d_in[3];
  const float* b2 = (const float*)d_in[4];
  float* out = (float*)d_out;

  char* ws = (char*)d_ws;
  float* h = (float*)ws;                 ws += (size_t)M * D2 * sizeof(float);
  signed char* qx = (signed char*)ws;    ws += (size_t)M * D2;
  signed char* qw = (signed char*)ws;    ws += (size_t)D2 * D1;
  float* sxb = (float*)ws;               ws += (size_t)M * sizeof(float);
  float* swb = (float*)ws;               ws += (size_t)D2 * sizeof(float);
  int* oidx  = (int*)ws;                 ws += (size_t)D2 * sizeof(int);
  int* n_out = (int*)ws;                 ws += 256;
  unsigned char* mask = (unsigned char*)ws;

  run_layer(x, W1, b1, h, M, D2, D1, qx, qw, sxb, swb, mask, oidx, n_out, stream);
  run_layer(h, W2, b2, out, M, D1, D2, qx, qw, sxb, swb, mask, oidx, n_out, stream);
}
